// StandardMoEBlock_26310969655468
// MI455X (gfx1250) — matrix operand order repaired and checked by reading, not compiled, not run on hardware
//
#include <hip/hip_runtime.h>

#define T_TOK 8192
#define DIM_D 1024
#define NUM_E 8
#define DIM_F 4096
#define BM 128
#define BN 128
#define BK 32
#define LDA 40   // LDS row stride (bf16 elems): 80B, 16B-aligned, conflict-free frag loads
#define LDB 40

typedef __attribute__((ext_vector_type(16))) __bf16 v16bf;
typedef __attribute__((ext_vector_type(8)))  __bf16 v8bf;
typedef __attribute__((ext_vector_type(4)))  __bf16 v4bf;
typedef __attribute__((ext_vector_type(2)))  __bf16 v2bf;
typedef __attribute__((ext_vector_type(8)))  float  v8f;

// ---------------- async global->LDS staging (CDNA5) ----------------
__device__ __forceinline__ void async_b128(unsigned lds_addr, const void* sbase, unsigned goff) {
  asm volatile("global_load_async_to_lds_b128 %0, %1, %2 offset:0"
               :: "v"(lds_addr), "v"(goff), "s"(sbase) : "memory");
}
__device__ __forceinline__ void wait_async0() {
  asm volatile("s_wait_asynccnt 0x0" ::: "memory");
}
__device__ __forceinline__ unsigned lds_off(const void* p) {
  return (unsigned)(size_t)p;
}

__device__ __forceinline__ v16bf join16(v8bf lo, v8bf hi) {
  return __builtin_shufflevector(lo, hi, 0,1,2,3,4,5,6,7,8,9,10,11,12,13,14,15);
}

// A fragment (16x32 bf16): lane<16 -> row M=lane, K 0..15; lane>=16 -> K 16..31
__device__ __forceinline__ v16bf load_a_frag(const __bf16* sA, int mrow, int lane) {
  int r = mrow + (lane & 15);
  int c = (lane < 16) ? 0 : 16;
  const __bf16* p = sA + r * LDA + c;
  v8bf lo = *(const v8bf*)p;
  v8bf hi = *(const v8bf*)(p + 8);
  return join16(lo, hi);
}

// B fragment (32x16 bf16), tile stored [n][k]: lane<16 -> col N=lane, K 0..15; lane>=16 -> K 16..31
__device__ __forceinline__ v16bf load_b_frag(const __bf16* sB, int ncol, int lane) {
  int n = ncol + (lane & 15);
  int c = (lane < 16) ? 0 : 16;
  const __bf16* p = sB + n * LDB + c;
  v8bf lo = *(const v8bf*)p;
  v8bf hi = *(const v8bf*)(p + 8);
  return join16(lo, hi);
}

#define WMMA_BF16(A, B, C) \
  __builtin_amdgcn_wmma_f32_16x16x32_bf16(false, (A), false, (B), (short)0, (C), false, false)

// ---------------- prep kernels ----------------
__global__ void zero_kernel(float* __restrict__ out, int n, int* __restrict__ counts) {
  int i = blockIdx.x * blockDim.x + threadIdx.x;
  if (i < n) out[i] = 0.f;
  if (blockIdx.x == 0 && threadIdx.x < 16) counts[threadIdx.x] = 0;
}

__global__ void convert_x_kernel(const float* __restrict__ x, __bf16* __restrict__ xbf, int n4) {
  int i = blockIdx.x * blockDim.x + threadIdx.x;
  if (i < n4) {
    float4 v = ((const float4*)x)[i];
    ((v4bf*)xbf)[i] = (v4bf){(__bf16)v.x, (__bf16)v.y, (__bf16)v.z, (__bf16)v.w};
  }
}

// src: [R][C] f32 (matrix z at z*R*C) -> dst: [C][R] bf16. Tiled transpose, 4B packed writes.
__global__ __launch_bounds__(256)
void transpose_convert(const float* __restrict__ src, __bf16* __restrict__ dst, int R, int C) {
  __shared__ float tile[32][33];
  size_t moff = (size_t)blockIdx.z * R * C;
  int c0 = blockIdx.x * 32, r0 = blockIdx.y * 32;
  int tx = threadIdx.x & 31, ty = threadIdx.x >> 5;  // 32x8
#pragma unroll
  for (int i = ty; i < 32; i += 8)
    tile[i][tx] = src[moff + (size_t)(r0 + i) * C + c0 + tx];
  __syncthreads();
  int tr = (tx & 15) * 2;
  int ib = ty + (tx >> 4) * 8;  // 0..15
#pragma unroll
  for (int rep = 0; rep < 2; rep++) {
    int i = ib + rep * 16;
    v2bf o = {(__bf16)tile[tr][i], (__bf16)tile[tr + 1][i]};
    *(v2bf*)(dst + moff + (size_t)(c0 + i) * R + r0 + tr) = o;
  }
}

// ---------------- router: one wave per token ----------------
__global__ __launch_bounds__(256)
void router_kernel(const float* __restrict__ x, const float* __restrict__ wr,
                   float* __restrict__ wfull, int* __restrict__ lists,
                   int* __restrict__ counts) {
  __shared__ float swr[DIM_D * NUM_E];
  for (int i = threadIdx.x; i < DIM_D * NUM_E; i += 256) swr[i] = wr[i];
  __syncthreads();
  int wave = threadIdx.x >> 5;
  int lane = threadIdx.x & 31;
  int tok = blockIdx.x * 8 + wave;
  float acc[NUM_E];
#pragma unroll
  for (int e = 0; e < NUM_E; e++) acc[e] = 0.f;
  const float* xr = x + (size_t)tok * DIM_D;
  for (int k = lane; k < DIM_D; k += 32) {
    float xv = xr[k];
#pragma unroll
    for (int e = 0; e < NUM_E; e++) acc[e] += xv * swr[k * NUM_E + e];
  }
#pragma unroll
  for (int e = 0; e < NUM_E; e++) {
#pragma unroll
    for (int off = 16; off > 0; off >>= 1) acc[e] += __shfl_xor(acc[e], off, 32);
  }
  if (lane == 0) {
    float m = acc[0];
#pragma unroll
    for (int e = 1; e < NUM_E; e++) m = fmaxf(m, acc[e]);
    float p[NUM_E]; float s = 0.f;
#pragma unroll
    for (int e = 0; e < NUM_E; e++) { p[e] = __expf(acc[e] - m); s += p[e]; }
#pragma unroll
    for (int e = 0; e < NUM_E; e++) p[e] /= s;
    int i0 = 0; float p0 = p[0];
#pragma unroll
    for (int e = 1; e < NUM_E; e++) if (p[e] > p0) { p0 = p[e]; i0 = e; }
    int i1 = -1; float p1 = -1.f;
#pragma unroll
    for (int e = 0; e < NUM_E; e++) if (e != i0 && p[e] > p1) { p1 = p[e]; i1 = e; }
    float inv = 1.f / (p0 + p1 + 1e-8f);
    float* wrow = wfull + (size_t)tok * 16;
#pragma unroll
    for (int j = 0; j < 16; j++) wrow[j] = 0.f;
    wrow[i0] = p0 * inv;
    wrow[i1] = p1 * inv;
    wrow[NUM_E] = 1.f;  // shared expert
    int pos0 = atomicAdd(&counts[i0], 1);
    lists[i0 * T_TOK + pos0] = tok;
    int pos1 = atomicAdd(&counts[i1], 1);
    lists[i1 * T_TOK + pos1] = tok;
    atomicAdd(&counts[NUM_E], 1);
    lists[NUM_E * T_TOK + tok] = tok;
  }
}

// ---------------- GEMM1: H[r,:] = silu(X[list[r]] @ Wg) * (X[list[r]] @ Wu) ----------------
__global__ __launch_bounds__(256)
void gemm1_gateup(const __bf16* __restrict__ xbf, const __bf16* __restrict__ wgt,
                  const __bf16* __restrict__ wut, const int* __restrict__ list,
                  const int* __restrict__ cntp, __bf16* __restrict__ H) {
  const int cnt = *cntp;
  const int my = blockIdx.y * BM;
  if (my >= cnt) return;
  const int N0 = blockIdx.x * BN;

  __shared__ __bf16 sA[2][BM * LDA] __attribute__((aligned(16)));
  __shared__ __bf16 sG[2][BN * LDB] __attribute__((aligned(16)));
  __shared__ __bf16 sU[2][BN * LDB] __attribute__((aligned(16)));
  __shared__ int sTok[BM];

  const int tid = threadIdx.x;
  if (tid < BM) {
    int r = my + tid;
    sTok[tid] = list[(r < cnt) ? r : (cnt - 1)];
  }
  __syncthreads();

  // Hoisted per-thread staging descriptors (invariant over k).
  const unsigned aOff = lds_off(&sA[0][0]);
  const unsigned gOff = lds_off(&sG[0][0]);
  const unsigned uOff = lds_off(&sU[0][0]);
  const unsigned ABUF = BM * LDA * 2, BBUF = BN * LDB * 2;
  const int r0s = tid >> 2, r1s = (tid + 256) >> 2;
  const int c0s = (tid & 3) * 8;
  const unsigned gA0 = ((unsigned)sTok[r0s] * DIM_D + c0s) * 2u;
  const unsigned gA1 = ((unsigned)sTok[r1s] * DIM_D + c0s) * 2u;
  const unsigned gB0 = ((unsigned)(N0 + r0s) * DIM_D + c0s) * 2u;
  const unsigned gB1 = ((unsigned)(N0 + r1s) * DIM_D + c0s) * 2u;
  const unsigned lA0 = aOff + (r0s * LDA + c0s) * 2u;
  const unsigned lA1 = aOff + (r1s * LDA + c0s) * 2u;
  const unsigned lB0 = (r0s * LDB + c0s) * 2u;
  const unsigned lB1 = (r1s * LDB + c0s) * 2u;

  auto stage = [&](int buf, int k0) {
    unsigned kb = (unsigned)k0 * 2u;
    unsigned ab = (unsigned)buf * ABUF, bb = (unsigned)buf * BBUF;
    async_b128(lA0 + ab, xbf, gA0 + kb);
    async_b128(lA1 + ab, xbf, gA1 + kb);
    async_b128(gOff + lB0 + bb, wgt, gB0 + kb);
    async_b128(gOff + lB1 + bb, wgt, gB1 + kb);
    async_b128(uOff + lB0 + bb, wut, gB0 + kb);
    async_b128(uOff + lB1 + bb, wut, gB1 + kb);
  };

  v8f accG[2][4], accU[2][4];
#pragma unroll
  for (int mt = 0; mt < 2; mt++)
#pragma unroll
    for (int nt = 0; nt < 4; nt++) {
      accG[mt][nt] = (v8f){0.f,0.f,0.f,0.f,0.f,0.f,0.f,0.f};
      accU[mt][nt] = (v8f){0.f,0.f,0.f,0.f,0.f,0.f,0.f,0.f};
    }

  const int wave = tid >> 5, lane = tid & 31;
  const int wm = (wave >> 1) * 32;
  const int wn = (wave & 1) * 64;

  auto compute = [&](int buf) {
    const __bf16* pA = sA[buf];
    const __bf16* pG = sG[buf];
    const __bf16* pU = sU[buf];
    v16bf a0 = load_a_frag(pA, wm, lane);
    v16bf a1 = load_a_frag(pA, wm + 16, lane);
    v16bf b0 = load_b_frag(pG, wn, lane);
    v16bf b1 = load_b_frag(pG, wn + 16, lane);
    v16bf b2 = load_b_frag(pG, wn + 32, lane);
    v16bf b3 = load_b_frag(pG, wn + 48, lane);
    accG[0][0] = WMMA_BF16(a0, b0, accG[0][0]);
    accG[1][0] = WMMA_BF16(a1, b0, accG[1][0]);
    accG[0][1] = WMMA_BF16(a0, b1, accG[0][1]);
    accG[1][1] = WMMA_BF16(a1, b1, accG[1][1]);
    accG[0][2] = WMMA_BF16(a0, b2, accG[0][2]);
    accG[1][2] = WMMA_BF16(a1, b2, accG[1][2]);
    accG[0][3] = WMMA_BF16(a0, b3, accG[0][3]);
    accG[1][3] = WMMA_BF16(a1, b3, accG[1][3]);
    b0 = load_b_frag(pU, wn, lane);
    b1 = load_b_frag(pU, wn + 16, lane);
    b2 = load_b_frag(pU, wn + 32, lane);
    b3 = load_b_frag(pU, wn + 48, lane);
    accU[0][0] = WMMA_BF16(a0, b0, accU[0][0]);
    accU[1][0] = WMMA_BF16(a1, b0, accU[1][0]);
    accU[0][1] = WMMA_BF16(a0, b1, accU[0][1]);
    accU[1][1] = WMMA_BF16(a1, b1, accU[1][1]);
    accU[0][2] = WMMA_BF16(a0, b2, accU[0][2]);
    accU[1][2] = WMMA_BF16(a1, b2, accU[1][2]);
    accU[0][3] = WMMA_BF16(a0, b3, accU[0][3]);
    accU[1][3] = WMMA_BF16(a1, b3, accU[1][3]);
  };

  stage(0, 0);
  wait_async0();
  __syncthreads();
  int buf = 0;
  const int KT = DIM_D / BK;
  for (int kt = 0; kt < KT - 1; kt++) {
    stage(buf ^ 1, (kt + 1) * BK);
    compute(buf);
    wait_async0();
    __syncthreads();
    buf ^= 1;
  }
  compute(buf);

#pragma unroll
  for (int mt = 0; mt < 2; mt++)
#pragma unroll
    for (int v = 0; v < 8; v++) {
      int M = wm + 16 * mt + v + ((lane < 16) ? 0 : 8);
      int row = my + M;
      if (row < cnt) {
        __bf16* hb = H + (size_t)row * DIM_F + N0 + wn + (lane & 15);
#pragma unroll
        for (int nt = 0; nt < 4; nt++) {
          float g = accG[mt][nt][v];
          float u = accU[mt][nt][v];
          hb[16 * nt] = (__bf16)((g / (1.f + __expf(-g))) * u);
        }
      }
    }
}

// ---------------- GEMM2: out[list[r],:] += wfull[list[r],ecol] * (H[r,:] @ Wd) ----------------
__global__ __launch_bounds__(256)
void gemm2_down(const __bf16* __restrict__ H, const __bf16* __restrict__ wdt,
                const int* __restrict__ list, const int* __restrict__ cntp,
                const float* __restrict__ wfull, int ecol, float* __restrict__ out) {
  const int cnt = *cntp;
  const int my = blockIdx.y * BM;
  if (my >= cnt) return;
  const int N0 = blockIdx.x * BN;  // over D

  __shared__ __bf16 sA[2][BM * LDA] __attribute__((aligned(16)));
  __shared__ __bf16 sB[2][BN * LDB] __attribute__((aligned(16)));
  __shared__ int sTok[BM];

  const int tid = threadIdx.x;
  if (tid < BM) {
    int r = my + tid;
    sTok[tid] = list[(r < cnt) ? r : (cnt - 1)];
  }
  __syncthreads();

  const unsigned aOff = lds_off(&sA[0][0]);
  const unsigned bOff = lds_off(&sB[0][0]);
  const unsigned ABUF = BM * LDA * 2, BBUF = BN * LDB * 2;
  const int r0s = tid >> 2, r1s = (tid + 256) >> 2;
  const int c0s = (tid & 3) * 8;
  int ra0 = my + r0s; if (ra0 >= cnt) ra0 = cnt - 1;
  int ra1 = my + r1s; if (ra1 >= cnt) ra1 = cnt - 1;
  const unsigned gA0 = ((unsigned)ra0 * DIM_F + c0s) * 2u;
  const unsigned gA1 = ((unsigned)ra1 * DIM_F + c0s) * 2u;
  const unsigned gB0 = ((unsigned)(N0 + r0s) * DIM_F + c0s) * 2u;
  const unsigned gB1 = ((unsigned)(N0 + r1s) * DIM_F + c0s) * 2u;
  const unsigned lA0 = aOff + (r0s * LDA + c0s) * 2u;
  const unsigned lA1 = aOff + (r1s * LDA + c0s) * 2u;
  const unsigned lB0 = bOff + (r0s * LDB + c0s) * 2u;
  const unsigned lB1 = bOff + (r1s * LDB + c0s) * 2u;

  auto stage = [&](int buf, int k0) {
    unsigned kb = (unsigned)k0 * 2u;
    unsigned ab = (unsigned)buf * ABUF, bb = (unsigned)buf * BBUF;
    async_b128(lA0 + ab, H, gA0 + kb);
    async_b128(lA1 + ab, H, gA1 + kb);
    async_b128(lB0 + bb, wdt, gB0 + kb);
    async_b128(lB1 + bb, wdt, gB1 + kb);
  };

  v8f acc[2][4];
#pragma unroll
  for (int mt = 0; mt < 2; mt++)
#pragma unroll
    for (int nt = 0; nt < 4; nt++)
      acc[mt][nt] = (v8f){0.f,0.f,0.f,0.f,0.f,0.f,0.f,0.f};

  const int wave = tid >> 5, lane = tid & 31;
  const int wm = (wave >> 1) * 32;
  const int wn = (wave & 1) * 64;

  auto compute = [&](int buf) {
    const __bf16* pA = sA[buf];
    const __bf16* pB = sB[buf];
    v16bf a0 = load_a_frag(pA, wm, lane);
    v16bf a1 = load_a_frag(pA, wm + 16, lane);
    v16bf b0 = load_b_frag(pB, wn, lane);
    v16bf b1 = load_b_frag(pB, wn + 16, lane);
    v16bf b2 = load_b_frag(pB, wn + 32, lane);
    v16bf b3 = load_b_frag(pB, wn + 48, lane);
    acc[0][0] = WMMA_BF16(a0, b0, acc[0][0]);
    acc[1][0] = WMMA_BF16(a1, b0, acc[1][0]);
    acc[0][1] = WMMA_BF16(a0, b1, acc[0][1]);
    acc[1][1] = WMMA_BF16(a1, b1, acc[1][1]);
    acc[0][2] = WMMA_BF16(a0, b2, acc[0][2]);
    acc[1][2] = WMMA_BF16(a1, b2, acc[1][2]);
    acc[0][3] = WMMA_BF16(a0, b3, acc[0][3]);
    acc[1][3] = WMMA_BF16(a1, b3, acc[1][3]);
  };

  stage(0, 0);
  wait_async0();
  __syncthreads();
  int buf = 0;
  const int KT = DIM_F / BK;
  for (int kt = 0; kt < KT - 1; kt++) {
    stage(buf ^ 1, (kt + 1) * BK);
    compute(buf);
    wait_async0();
    __syncthreads();
    buf ^= 1;
  }
  compute(buf);

#pragma unroll
  for (int mt = 0; mt < 2; mt++)
#pragma unroll
    for (int v = 0; v < 8; v++) {
      int M = wm + 16 * mt + v + ((lane < 16) ? 0 : 8);
      int row = my + M;
      if (row < cnt) {
        int tok = sTok[M];
        float w = wfull[(size_t)tok * 16 + ecol];
        float* ob = out + (size_t)tok * DIM_D + N0 + wn + (lane & 15);
#pragma unroll
        for (int nt = 0; nt < 4; nt++)
          unsafeAtomicAdd(ob + 16 * nt, w * acc[mt][nt][v]);
      }
    }
}

extern "C" void kernel_launch(void* const* d_in, const int* in_sizes, int n_in,
                              void* d_out, int out_size, void* d_ws, size_t ws_size,
                              hipStream_t stream) {
  const float* x        = (const float*)d_in[0];
  const float* w_router = (const float*)d_in[1];
  const float* w_gate   = (const float*)d_in[2];
  const float* w_up     = (const float*)d_in[3];
  const float* w_down   = (const float*)d_in[4];
  const float* wsg      = (const float*)d_in[5];
  const float* wsu      = (const float*)d_in[6];
  const float* wsd      = (const float*)d_in[7];
  float* out = (float*)d_out;

  const size_t DF = (size_t)DIM_D * DIM_F;
  char* ws = (char*)d_ws;
  float*  wfull  = (float*)(ws);              // 512 KB
  int*    lists  = (int*)(ws + 0x80000);      // 512 KB
  int*    counts = (int*)(ws + 0x100000);     // 4 KB slot
  __bf16* xbf    = (__bf16*)(ws + 0x101000);  // 16 MB
  __bf16* Hbuf   = (__bf16*)(ws + 0x1101000); // 64 MB
  __bf16* wgT    = (__bf16*)(ws + 0x5101000); // 9 * 8 MB  [F][D]
  __bf16* wuT    = wgT + 9 * DF;              // 9 * 8 MB  [F][D]
  __bf16* wdT    = wuT + 9 * DF;              // 9 * 8 MB  [D][F]

  zero_kernel<<<dim3((T_TOK * DIM_D + 255) / 256), dim3(256), 0, stream>>>(out, T_TOK * DIM_D, counts);
  convert_x_kernel<<<dim3(T_TOK * DIM_D / 4 / 256), dim3(256), 0, stream>>>(x, xbf, T_TOK * DIM_D / 4);

  dim3 tb(256);
  transpose_convert<<<dim3(DIM_F / 32, DIM_D / 32, NUM_E), tb, 0, stream>>>(w_gate, wgT, DIM_D, DIM_F);
  transpose_convert<<<dim3(DIM_F / 32, DIM_D / 32, NUM_E), tb, 0, stream>>>(w_up,   wuT, DIM_D, DIM_F);
  transpose_convert<<<dim3(DIM_D / 32, DIM_F / 32, NUM_E), tb, 0, stream>>>(w_down, wdT, DIM_F, DIM_D);
  transpose_convert<<<dim3(DIM_F / 32, DIM_D / 32, 1), tb, 0, stream>>>(wsg, wgT + 8 * DF, DIM_D, DIM_F);
  transpose_convert<<<dim3(DIM_F / 32, DIM_D / 32, 1), tb, 0, stream>>>(wsu, wuT + 8 * DF, DIM_D, DIM_F);
  transpose_convert<<<dim3(DIM_D / 32, DIM_F / 32, 1), tb, 0, stream>>>(wsd, wdT + 8 * DF, DIM_F, DIM_D);

  router_kernel<<<dim3(T_TOK / 8), dim3(256), 0, stream>>>(x, w_router, wfull, lists, counts);

  dim3 g1(DIM_F / BN, T_TOK / BM);
  dim3 g2(DIM_D / BN, T_TOK / BM);
  for (int e = 0; e < NUM_E + 1; e++) {
    gemm1_gateup<<<g1, dim3(256), 0, stream>>>(xbf, wgT + (size_t)e * DF, wuT + (size_t)e * DF,
                                               lists + (size_t)e * T_TOK, counts + e, Hbuf);
    gemm2_down <<<g2, dim3(256), 0, stream>>>(Hbuf, wdT + (size_t)e * DF,
                                              lists + (size_t)e * T_TOK, counts + e, wfull, e, out);
  }
}
